// SelfAttention_5239860101728
// MI455X (gfx1250) — matrix operand order repaired and checked by reading, not compiled, not run on hardware
//
#include <hip/hip_runtime.h>
#include <hip/hip_bf16.h>

// ---------------------------------------------------------------------------
// Self-attention forward for MI455X (gfx1250), wave32 WMMA bf16 pipeline.
//   q = x Wq^T + bq ; k = x Wk^T + bk ; v = x Wv^T + bv
//   out = softmax(q k^T / sqrt(E)) v
// B=16, S=2048, E=1024 (all static).
//
// Pipeline (all bf16 matrix math on v_wmma_f32_16x16x32_bf16, f32 accumulate):
//   k0a: x (f32)            -> xb  bf16 [32768][1024]
//   k0b: Wq|Wk|Wv, biases   -> Wcat bf16 [3072][1024], bcat f32 [3072]
//   k1 : qkv = xb Wcat^T + bcat      (bf16 GEMM, 4x4 frags/wave)
//   k2 : vT[b][e][t] = v[b][t][e]
//   k3a: scores = (q k^T) / 32  -> f32   (bf16 GEMM, 4x4 frags/wave)
//   k3b: probs = softmax_rows(scores) -> bf16
//   k4 : out = probs @ v  (A=probs, B=vT)  -> f32  (4x4 frags/wave)
// ---------------------------------------------------------------------------

typedef __bf16 bf16_t;
typedef __attribute__((ext_vector_type(16))) __bf16 v16bf;
typedef __attribute__((ext_vector_type(8)))  __bf16 v8bf;
typedef __attribute__((ext_vector_type(8)))  float  v8f;
typedef __attribute__((ext_vector_type(4)))  float  v4f;

#define ATT_B 16
#define ATT_S 2048
#define ATT_E 1024
#define ATT_3E 3072

static __device__ inline v8f zero8() {
    v8f z;
#pragma unroll
    for (int i = 0; i < 8; ++i) z[i] = 0.0f;
    return z;
}

static __device__ inline v16bf cat8(v8bf lo, v8bf hi) {
    v16bf r;
#pragma unroll
    for (int i = 0; i < 8; ++i) { r[i] = lo[i]; r[8 + i] = hi[i]; }
    return r;
}

// A fragment (16x32 bf16, row-major source, ld elements per row).
// ISA 7.12.2: lanes 0-15 hold K {0..7,16..23}, lanes 16-31 {8..15,24..31}
// for row M = lane&15  ->  two contiguous 16-byte runs per lane.
static __device__ inline v16bf load_a_bf16(const bf16_t* __restrict__ src,
                                           int ld, int row0, int k0, int lane) {
    const int half = lane >> 4, r = lane & 15;
    const bf16_t* p = src + (size_t)(row0 + r) * ld + k0 + (half ? 8 : 0);
    v8bf lo = *(const v8bf*)p;
    v8bf hi = *(const v8bf*)(p + 16);
    return cat8(lo, hi);
}

// B fragment (32x16 bf16) where B[K][N] = src[N][K] (src row-major, ld).
// Lane (half,col) holds K {8*half+0..7, 16+8*half+0..7} for column
// N = lane&15 -> two contiguous 16-byte runs per lane.
static __device__ inline v16bf load_bt_bf16(const bf16_t* __restrict__ src,
                                            int ld, int col0, int k0, int lane) {
    const int half = lane >> 4, c = lane & 15;
    const bf16_t* p = src + (size_t)(col0 + c) * ld + k0 + (half ? 8 : 0);
    v8bf lo = *(const v8bf*)p;
    v8bf hi = *(const v8bf*)(p + 16);
    return cat8(lo, hi);
}

static __device__ inline v8f wmma_bf16(v16bf a, v16bf b, v8f c) {
    // D = A x B + C  (16x16x32, f32 accumulate)
    return __builtin_amdgcn_wmma_f32_16x16x32_bf16(
        /*neg_a=*/false, a, /*neg_b=*/false, b,
        /*c_mod=*/(short)0, c, /*reuse_a=*/false, /*reuse_b=*/false);
}

static __device__ inline float wave32_max(float v) {
#pragma unroll
    for (int m = 16; m >= 1; m >>= 1) v = fmaxf(v, __shfl_xor(v, m, 32));
    return v;
}
static __device__ inline float wave32_sum(float v) {
#pragma unroll
    for (int m = 16; m >= 1; m >>= 1) v += __shfl_xor(v, m, 32);
    return v;
}

// ---------------------------------------------------------------------------
// k0a: convert x (f32) -> xb (bf16), 8 elements per thread.
// ---------------------------------------------------------------------------
__global__ __launch_bounds__(256)
void convert_x_kernel(const float* __restrict__ x, bf16_t* __restrict__ xb) {
    const size_t base = ((size_t)blockIdx.x * 256 + threadIdx.x) * 8;
    v4f f0 = *(const v4f*)(x + base);
    v4f f1 = *(const v4f*)(x + base + 4);
    v8bf o;
#pragma unroll
    for (int i = 0; i < 4; ++i) { o[i] = (bf16_t)f0[i]; o[4 + i] = (bf16_t)f1[i]; }
    *(v8bf*)(xb + base) = o;
}

// ---------------------------------------------------------------------------
// k0b: pack Wq|Wk|Wv -> Wcat bf16 [3072][1024]; bq|bk|bv -> bcat f32 [3072].
// ---------------------------------------------------------------------------
__global__ __launch_bounds__(256)
void pack_w_kernel(const float* __restrict__ Wq, const float* __restrict__ bq,
                   const float* __restrict__ Wk, const float* __restrict__ bk,
                   const float* __restrict__ Wv, const float* __restrict__ bv,
                   bf16_t* __restrict__ Wcat, float* __restrict__ bcat) {
    const unsigned tid = blockIdx.x * 256u + threadIdx.x;     // < 393216
    const size_t base = (size_t)tid * 8;                      // element index
    const int mat = (int)(base >> 20);                        // 0:q 1:k 2:v
    const size_t rem = base & ((1u << 20) - 1);
    const float* W = (mat == 0) ? Wq : (mat == 1) ? Wk : Wv;
    v4f f0 = *(const v4f*)(W + rem);
    v4f f1 = *(const v4f*)(W + rem + 4);
    v8bf o;
#pragma unroll
    for (int i = 0; i < 4; ++i) { o[i] = (bf16_t)f0[i]; o[4 + i] = (bf16_t)f1[i]; }
    *(v8bf*)(Wcat + base) = o;

    if (tid < ATT_3E) {
        const int m2 = tid >> 10, r2 = tid & 1023;
        bcat[tid] = (m2 == 0) ? bq[r2] : (m2 == 1) ? bk[r2] : bv[r2];
    }
}

// ---------------------------------------------------------------------------
// Generic epilogue helper: C fragment (VGPR j holds rows M=j / M=j+8).
// ---------------------------------------------------------------------------

// k1: qkv GEMM.  M=32768, N=3072, K=1024.  Block 128x256, 8 waves = 2x4,
// wave tile 64x64 (4x4 fragments).
__global__ __launch_bounds__(256)
void qkv_gemm_kernel(const bf16_t* __restrict__ xb,
                     const bf16_t* __restrict__ Wcat,
                     const float* __restrict__ bcat,
                     bf16_t* __restrict__ qkv) {
    const int lane = threadIdx.x & 31;
    const int wave = threadIdx.x >> 5;
    const int half = lane >> 4;
    const int col  = lane & 15;

    const int n0 = blockIdx.x * 256;
    const int m0 = blockIdx.y * 128;
    const int wm = wave >> 2;            // 0..1
    const int wn = wave & 3;             // 0..3
    const int rowA0 = m0 + wm * 64;
    const int colB0 = n0 + wn * 64;

    v8f acc[4][4];
#pragma unroll
    for (int i = 0; i < 4; ++i)
#pragma unroll
        for (int j = 0; j < 4; ++j) acc[i][j] = zero8();

#pragma unroll 1
    for (int k0 = 0; k0 < ATT_E; k0 += 32) {
        if (k0 + 32 < ATT_E) {
            __builtin_prefetch(xb + (size_t)(rowA0 + (lane & 15)) * ATT_E + k0 + 32, 0, 1);
            __builtin_prefetch(Wcat + (size_t)(colB0 + (lane & 15)) * ATT_E + k0 + 32, 0, 1);
        }
        v16bf a[4], b[4];
#pragma unroll
        for (int i = 0; i < 4; ++i)
            a[i] = load_a_bf16(xb, ATT_E, rowA0 + i * 16, k0, lane);
#pragma unroll
        for (int j = 0; j < 4; ++j)
            b[j] = load_bt_bf16(Wcat, ATT_E, colB0 + j * 16, k0, lane);
#pragma unroll
        for (int i = 0; i < 4; ++i)
#pragma unroll
            for (int j = 0; j < 4; ++j)
                acc[i][j] = wmma_bf16(a[i], b[j], acc[i][j]);
    }

#pragma unroll
    for (int i = 0; i < 4; ++i) {
#pragma unroll
        for (int j = 0; j < 4; ++j) {
            const int ccol = colB0 + j * 16 + col;
            const float bb = bcat[ccol];
#pragma unroll
            for (int jj = 0; jj < 8; ++jj) {
                const int row = rowA0 + i * 16 + jj + 8 * half;
                qkv[(size_t)row * ATT_3E + ccol] = (bf16_t)(acc[i][j][jj] + bb);
            }
        }
    }
}

// ---------------------------------------------------------------------------
// k2: vT[b][e][t] = v[b][t][e]
// ---------------------------------------------------------------------------
__global__ __launch_bounds__(256)
void transpose_v_kernel(const bf16_t* __restrict__ qkv,
                        bf16_t* __restrict__ vT) {
    const unsigned idx = blockIdx.x * 256u + threadIdx.x;   // < B*S*E
    const int e = idx & (ATT_E - 1);
    const int t = (idx >> 10) & (ATT_S - 1);
    const int b = idx >> 21;
    vT[((size_t)b * ATT_E + e) * ATT_S + t] =
        qkv[((size_t)(b * ATT_S + t)) * ATT_3E + 2048 + e];
}

// ---------------------------------------------------------------------------
// k3a: scores GEMM.  Per batch M=2048, N=2048, K=1024; f32 scores, scaled.
// Block 128x256, wave tile 64x64 (4x4 fragments).
// ---------------------------------------------------------------------------
__global__ __launch_bounds__(256)
void scores_gemm_kernel(const bf16_t* __restrict__ qkv,
                        float* __restrict__ scores) {
    const int lane = threadIdx.x & 31;
    const int wave = threadIdx.x >> 5;
    const int half = lane >> 4;
    const int col  = lane & 15;

    const int n0  = blockIdx.x * 256;       // key-column block (0..2047)
    const int mg0 = blockIdx.y * 128;       // global query row block
    const int b   = mg0 >> 11;
    const int s0  = mg0 & (ATT_S - 1);
    const int wm = wave >> 2;
    const int wn = wave & 3;
    const int rowA0 = s0 + wm * 64;
    const int colB0 = n0 + wn * 64;

    const bf16_t* qbase = qkv + (size_t)(b * ATT_S) * ATT_3E;          // q, ld 3E
    const bf16_t* kbase = qkv + (size_t)(b * ATT_S) * ATT_3E + 1024;   // k, ld 3E

    v8f acc[4][4];
#pragma unroll
    for (int i = 0; i < 4; ++i)
#pragma unroll
        for (int j = 0; j < 4; ++j) acc[i][j] = zero8();

#pragma unroll 1
    for (int e0 = 0; e0 < ATT_E; e0 += 32) {
        if (e0 + 32 < ATT_E) {
            __builtin_prefetch(qbase + (size_t)(rowA0 + (lane & 15)) * ATT_3E + e0 + 32, 0, 1);
            __builtin_prefetch(kbase + (size_t)(colB0 + (lane & 15)) * ATT_3E + e0 + 32, 0, 1);
        }
        v16bf a[4], bfr[4];
#pragma unroll
        for (int i = 0; i < 4; ++i)
            a[i] = load_a_bf16(qbase, ATT_3E, rowA0 + i * 16, e0, lane);
#pragma unroll
        for (int j = 0; j < 4; ++j)
            bfr[j] = load_bt_bf16(kbase, ATT_3E, colB0 + j * 16, e0, lane);
#pragma unroll
        for (int i = 0; i < 4; ++i)
#pragma unroll
            for (int j = 0; j < 4; ++j)
                acc[i][j] = wmma_bf16(a[i], bfr[j], acc[i][j]);
    }

    const float scale = 0.03125f;           // 1/sqrt(1024)
    float* sbase = scores + (size_t)b * ATT_S * ATT_S;
#pragma unroll
    for (int i = 0; i < 4; ++i)
#pragma unroll
        for (int j = 0; j < 4; ++j)
#pragma unroll
            for (int jj = 0; jj < 8; ++jj) {
                const int row = rowA0 + i * 16 + jj + 8 * half;
                const int t   = colB0 + j * 16 + col;
                sbase[(size_t)row * ATT_S + t] = acc[i][j][jj] * scale;
            }
}

// ---------------------------------------------------------------------------
// k3b: row softmax: probs[row] = softmax(scores[row]), bf16 out.
// One block (256 threads) per row of 2048; 8 f32 per thread.
// ---------------------------------------------------------------------------
__global__ __launch_bounds__(256)
void softmax_kernel(const float* __restrict__ scores,
                    bf16_t* __restrict__ probs) {
    const size_t row = blockIdx.x;                       // < B*S
    const float* src = scores + row * ATT_S;
    bf16_t* dst = probs + row * ATT_S;
    const int t0 = threadIdx.x * 8;
    const int wave = threadIdx.x >> 5;
    const int lane = threadIdx.x & 31;

    float v[8];
    {
        v4f f0 = *(const v4f*)(src + t0);
        v4f f1 = *(const v4f*)(src + t0 + 4);
#pragma unroll
        for (int i = 0; i < 4; ++i) { v[i] = f0[i]; v[4 + i] = f1[i]; }
    }

    __shared__ float red[8];
    __shared__ float bval[1];

    // global max
    float m = v[0];
#pragma unroll
    for (int i = 1; i < 8; ++i) m = fmaxf(m, v[i]);
    m = wave32_max(m);
    if (lane == 0) red[wave] = m;
    __syncthreads();
    if (threadIdx.x == 0) {
        float g = red[0];
#pragma unroll
        for (int w = 1; w < 8; ++w) g = fmaxf(g, red[w]);
        bval[0] = g;
    }
    __syncthreads();
    const float gmax = bval[0];

    // exp + global sum
    float s = 0.0f;
#pragma unroll
    for (int i = 0; i < 8; ++i) { v[i] = __expf(v[i] - gmax); s += v[i]; }
    s = wave32_sum(s);
    __syncthreads();                 // red[] reuse
    if (lane == 0) red[wave] = s;
    __syncthreads();
    if (threadIdx.x == 0) {
        float g = 0.0f;
#pragma unroll
        for (int w = 0; w < 8; ++w) g += red[w];
        bval[0] = 1.0f / g;
    }
    __syncthreads();
    const float rinv = bval[0];

    v8bf o;
#pragma unroll
    for (int i = 0; i < 8; ++i) o[i] = (bf16_t)(v[i] * rinv);
    *(v8bf*)(dst + t0) = o;
}

// ---------------------------------------------------------------------------
// k4: out = probs @ v.  Per batch M=2048, N=1024, K=2048.
// Block 128x256, wave tile 64x64 (4x4 fragments).  f32 output.
// ---------------------------------------------------------------------------
__global__ __launch_bounds__(256)
void out_gemm_kernel(const bf16_t* __restrict__ probs,
                     const bf16_t* __restrict__ vT,
                     float* __restrict__ out) {
    const int lane = threadIdx.x & 31;
    const int wave = threadIdx.x >> 5;
    const int half = lane >> 4;
    const int col  = lane & 15;

    const int n0  = blockIdx.x * 256;      // output feature block (0..1023)
    const int mg0 = blockIdx.y * 128;      // global row block
    const int b   = mg0 >> 11;
    const int s0  = mg0 & (ATT_S - 1);
    const int wm = wave >> 2;
    const int wn = wave & 3;
    const int rowA0 = s0 + wm * 64;
    const int colB0 = n0 + wn * 64;

    const bf16_t* A  = probs + (size_t)b * ATT_S * ATT_S;   // [s][t], ld=S
    const bf16_t* Bv = vT + (size_t)b * ATT_E * ATT_S;      // [e][t], ld=S

    v8f acc[4][4];
#pragma unroll
    for (int i = 0; i < 4; ++i)
#pragma unroll
        for (int j = 0; j < 4; ++j) acc[i][j] = zero8();

#pragma unroll 1
    for (int t0 = 0; t0 < ATT_S; t0 += 32) {
        if (t0 + 32 < ATT_S) {
            __builtin_prefetch(A + (size_t)(rowA0 + (lane & 15)) * ATT_S + t0 + 32, 0, 1);
            __builtin_prefetch(Bv + (size_t)(colB0 + (lane & 15)) * ATT_S + t0 + 32, 0, 1);
        }
        v16bf a[4], bfr[4];
#pragma unroll
        for (int i = 0; i < 4; ++i)
            a[i] = load_a_bf16(A, ATT_S, rowA0 + i * 16, t0, lane);
#pragma unroll
        for (int j = 0; j < 4; ++j)
            bfr[j] = load_bt_bf16(Bv, ATT_S, colB0 + j * 16, t0, lane);
#pragma unroll
        for (int i = 0; i < 4; ++i)
#pragma unroll
            for (int j = 0; j < 4; ++j)
                acc[i][j] = wmma_bf16(a[i], bfr[j], acc[i][j]);
    }

#pragma unroll
    for (int i = 0; i < 4; ++i)
#pragma unroll
        for (int j = 0; j < 4; ++j)
#pragma unroll
            for (int jj = 0; jj < 8; ++jj) {
                const int srow = rowA0 + i * 16 + jj + 8 * half;
                const int ccol = colB0 + j * 16 + col;
                out[((size_t)(b * ATT_S + srow)) * ATT_E + ccol] = acc[i][j][jj];
            }
}

// ---------------------------------------------------------------------------
// Host entry
// ---------------------------------------------------------------------------
extern "C" void kernel_launch(void* const* d_in, const int* in_sizes, int n_in,
                              void* d_out, int out_size, void* d_ws, size_t ws_size,
                              hipStream_t stream) {
    const float* x  = (const float*)d_in[0];
    const float* Wq = (const float*)d_in[1];
    const float* bq = (const float*)d_in[2];
    const float* Wk = (const float*)d_in[3];
    const float* bk = (const float*)d_in[4];
    const float* Wv = (const float*)d_in[5];
    const float* bv = (const float*)d_in[6];
    float* out = (float*)d_out;

    // Workspace layout (lifetimes allow xb/scores overlay):
    //   qkv   bf16 [B][S][3E]  192 MB      (k1 .. k4)
    //   vT    bf16 [B][E][S]    64 MB      (k2 .. k4)
    //   Wcat  bf16 [3E][E]       6 MB      (k0b .. k1)
    //   bcat  f32  [3E]         16 KB
    //   probs bf16 [B][S][S]   128 MB      (k3b .. k4)
    //   region: xb bf16 64 MB (k0a..k1)  OVERLAYS  scores f32 256 MB (k3a..k3b)
    char* ws = (char*)d_ws;
    bf16_t* qkv   = (bf16_t*)ws;                 ws += (size_t)ATT_B * ATT_S * ATT_3E * 2;
    bf16_t* vT    = (bf16_t*)ws;                 ws += (size_t)ATT_B * ATT_E * ATT_S * 2;
    bf16_t* Wcat  = (bf16_t*)ws;                 ws += (size_t)ATT_3E * ATT_E * 2;
    float*  bcat  = (float*)ws;                  ws += 16384;
    bf16_t* probs = (bf16_t*)ws;                 ws += (size_t)ATT_B * ATT_S * ATT_S * 2;
    bf16_t* xb     = (bf16_t*)ws;                // first 64 MB of region
    float*  scores = (float*)ws;                 // full 256 MB region

    // k0a: x -> bf16
    convert_x_kernel<<<((size_t)ATT_B * ATT_S * ATT_E) / (256 * 8), 256, 0, stream>>>(x, xb);
    // k0b: pack weights + biases
    pack_w_kernel<<<((size_t)3 * ATT_E * ATT_E) / (256 * 8), 256, 0, stream>>>(
        Wq, bq, Wk, bk, Wv, bv, Wcat, bcat);
    // k1: fused QKV projection
    qkv_gemm_kernel<<<dim3(ATT_3E / 256, (ATT_B * ATT_S) / 128), 256, 0, stream>>>(
        xb, Wcat, bcat, qkv);
    // k2: transpose v -> vT
    transpose_v_kernel<<<((unsigned)ATT_B * ATT_S * ATT_E) / 256, 256, 0, stream>>>(qkv, vT);
    // k3a: scaled scores (f32)
    scores_gemm_kernel<<<dim3(ATT_S / 256, (ATT_B * ATT_S) / 128), 256, 0, stream>>>(
        qkv, scores);
    // k3b: row softmax -> probs (bf16)
    softmax_kernel<<<ATT_B * ATT_S, 256, 0, stream>>>(scores, probs);
    // k4: out = probs @ v
    out_gemm_kernel<<<dim3(ATT_E / 256, (ATT_B * ATT_S) / 128), 256, 0, stream>>>(
        probs, vT, out);
}
